// _DistributedAttention_25812753449379
// MI455X (gfx1250) — matrix operand order repaired and checked by reading, not compiled, not run on hardware
//
#include <hip/hip_runtime.h>
#include <hip/hip_bf16.h>

// ---------------------------------------------------------------------------
// DistributedAttention for MI455X (gfx1250, wave32, WMMA).
//
// Pipeline (all launches stream-ordered, graph-capture safe):
//   1) global min/max of q,k,v (two-stage deterministic reduction)
//   2) int4 quant-dequant (faithful uint8 nibble pack/unpack semantics) +
//      all-to-all permutation + f16 conversion:
//        Qh,Kh: [b][h][t=2048][d=128] row-major f16
//        Vt   : [b][h][d=128][t=2048] transposed f16 (so PV B-fragments are
//               contiguous per-lane 32B loads)
//   3) flash attention, one wave32 per 16-query block, v_wmma_f32_16x16x32_f16
//      for both QK^T and PV, f32 online softmax, -> ctx f32 [t][b][h*128]
//   4) global min/max of ctx
//   5) quant-dequant of ctx + reverse a2a -> out f32 [512][2][4096]
//
// Workspace layout (needs ~40 MiB + 8.1 KiB):
//   [0,8Mi)    Qh f16      [8Mi,16Mi)  Kh f16     [16Mi,24Mi) Vt f16
//   [24Mi,40Mi) ctx f32
//   [40Mi, +32) 4 x {mn,step} result slots
//   [40Mi+64, +8Ki) per-block min/max partials
// ---------------------------------------------------------------------------

typedef _Float16 v16h __attribute__((ext_vector_type(16)));
typedef _Float16 v8h  __attribute__((ext_vector_type(8)));
typedef float    v8f  __attribute__((ext_vector_type(8)));

constexpr int S_ = 512, B_ = 2, H_ = 32, D_ = 128, W_ = 4, HH_ = 8, T_ = 2048;
constexpr int HD_ = HH_ * D_;                 // 1024
constexpr int N_  = S_ * B_ * H_ * D_;        // 4,194,304 elements (q/k/v and ctx)
constexpr int NPART_ = 1024;

constexpr size_t QH_OFF  = 0;
constexpr size_t KH_OFF  = (size_t)8  << 20;
constexpr size_t VT_OFF  = (size_t)16 << 20;
constexpr size_t CTX_OFF = (size_t)24 << 20;
constexpr size_t RES_OFF = (size_t)40 << 20;
constexpr size_t PART_OFF = RES_OFF + 64;

// ---------------- min/max reduction (deterministic, two stage) --------------

__global__ void minmax_part_kernel(const float* __restrict__ x, int n,
                                   float* __restrict__ pmin, float* __restrict__ pmax) {
  __shared__ float smin[256], smax[256];
  float mn = __builtin_inff(), mx = -__builtin_inff();
  for (int i = blockIdx.x * 256 + threadIdx.x; i < n; i += gridDim.x * 256) {
    float v = x[i];
    mn = fminf(mn, v);
    mx = fmaxf(mx, v);
  }
  smin[threadIdx.x] = mn; smax[threadIdx.x] = mx;
  __syncthreads();
  for (int off = 128; off > 0; off >>= 1) {
    if ((int)threadIdx.x < off) {
      smin[threadIdx.x] = fminf(smin[threadIdx.x], smin[threadIdx.x + off]);
      smax[threadIdx.x] = fmaxf(smax[threadIdx.x], smax[threadIdx.x + off]);
    }
    __syncthreads();
  }
  if (threadIdx.x == 0) { pmin[blockIdx.x] = smin[0]; pmax[blockIdx.x] = smax[0]; }
}

__global__ void minmax_final_kernel(const float* __restrict__ pmin,
                                    const float* __restrict__ pmax,
                                    int nparts, float* __restrict__ ms) {
  __shared__ float smin[256], smax[256];
  float mn = __builtin_inff(), mx = -__builtin_inff();
  for (int i = threadIdx.x; i < nparts; i += 256) {
    mn = fminf(mn, pmin[i]);
    mx = fmaxf(mx, pmax[i]);
  }
  smin[threadIdx.x] = mn; smax[threadIdx.x] = mx;
  __syncthreads();
  for (int off = 128; off > 0; off >>= 1) {
    if ((int)threadIdx.x < off) {
      smin[threadIdx.x] = fminf(smin[threadIdx.x], smin[threadIdx.x + off]);
      smax[threadIdx.x] = fmaxf(smax[threadIdx.x], smax[threadIdx.x + off]);
    }
    __syncthreads();
  }
  if (threadIdx.x == 0) {
    ms[0] = smin[0];
    ms[1] = (smax[0] - smin[0]) / 16.0f;   // step
  }
}

// ---------------- int4 quant-dequant (faithful uint8 nibble semantics) ------

__device__ __forceinline__ void qdq_pair(float xhi, float xlo, float mn, float step,
                                         float& out_hi, float& out_lo) {
  // q = floor((x - mn + step/2)/step) cast to uint8 (values 0..16)
  unsigned qh = (unsigned)(unsigned char)floorf((xhi - mn + 0.5f * step) / step);
  unsigned ql = (unsigned)(unsigned char)floorf((xlo - mn + 0.5f * step) / step);
  unsigned packed = ((qh << 4) + ql) & 0xFFu;   // uint8 wrap: q==16 carries/overflows
  out_hi = (float)(packed >> 4) * step + mn;
  out_lo = (float)(packed & 0xFu) * step + mn;
}

// q/k: dequant + a2a -> f16 [b][hh][t][128]
__global__ void qk_transform_kernel(const float* __restrict__ x, const float* __restrict__ ms,
                                    _Float16* __restrict__ dst) {
  int u = blockIdx.x * 256 + threadIdx.x;           // one thread per (d, d+64) pair
  int d = u & 63; int r = u >> 6;
  int Hi = r & (H_ - 1); r >>= 5;
  int b = r & (B_ - 1); int s = r >> 1;
  float mn = ms[0], step = ms[1];
  const float* row = x + (((size_t)s * B_ + b) * H_ + Hi) * D_;
  float hi, lo;
  qdq_pair(row[d], row[d + 64], mn, step, hi, lo);
  int w = Hi >> 3, hh = Hi & 7;
  int t = w * S_ + s;
  _Float16* orow = dst + ((size_t)(b * HH_ + hh) * T_ + t) * D_;
  orow[d]      = (_Float16)hi;
  orow[d + 64] = (_Float16)lo;
}

// v: dequant + a2a -> transposed f16 [b][hh][128][t]
__global__ void v_transform_kernel(const float* __restrict__ x, const float* __restrict__ ms,
                                   _Float16* __restrict__ dst) {
  int u = blockIdx.x * 256 + threadIdx.x;
  int d = u & 63; int r = u >> 6;
  int Hi = r & (H_ - 1); r >>= 5;
  int b = r & (B_ - 1); int s = r >> 1;
  float mn = ms[0], step = ms[1];
  const float* row = x + (((size_t)s * B_ + b) * H_ + Hi) * D_;
  float hi, lo;
  qdq_pair(row[d], row[d + 64], mn, step, hi, lo);
  int w = Hi >> 3, hh = Hi & 7;
  int t = w * S_ + s;
  _Float16* base = dst + (size_t)(b * HH_ + hh) * D_ * T_;
  base[(size_t)d * T_ + t]        = (_Float16)hi;
  base[(size_t)(d + 64) * T_ + t] = (_Float16)lo;
}

// ---------------- flash attention core (wave32 + WMMA) ----------------------

__global__ __launch_bounds__(32) void flash_attn_kernel(
    const _Float16* __restrict__ Qh, const _Float16* __restrict__ Kh,
    const _Float16* __restrict__ Vt, float* __restrict__ ctx) {
  const int lane = threadIdx.x;      // 0..31
  const int hl   = lane >> 4;        // lane half (WMMA layout)
  const int ln16 = lane & 15;

  int blk = blockIdx.x;              // (b, hh, rowblk)
  const int rowblk = blk & 127; blk >>= 7;
  const int hh = blk & 7;       blk >>= 3;
  const int b  = blk;

  const _Float16* Qb = Qh + (size_t)(b * HH_ + hh) * T_ * D_;
  const _Float16* Kb = Kh + (size_t)(b * HH_ + hh) * T_ * D_;
  const _Float16* Vb = Vt + (size_t)(b * HH_ + hh) * D_ * T_;

  // Q A-fragments for this 16-row block: 4 chunks of K=32.
  // A 16x32 layout: lane holds row (lane&15); elem e -> d = hl*16+e,
  // all within the chunk.
  const _Float16* qrow = Qb + (size_t)(rowblk * 16 + ln16) * D_;
  v16h qa[4];
#pragma unroll
  for (int c = 0; c < 4; ++c) {
    qa[c] = *(const v16h*)(qrow + c * 32 + hl * 16);
  }

  v8f acc[8];
  float m[8], l[8];
#pragma unroll
  for (int i = 0; i < 8; ++i) { m[i] = -__builtin_inff(); l[i] = 0.0f; }
#pragma unroll
  for (int dt = 0; dt < 8; ++dt)
#pragma unroll
    for (int i = 0; i < 8; ++i) acc[dt][i] = 0.0f;

  __shared__ _Float16 Pl[16][32];
  const float scale = 0.08838834764831845f;   // 1/sqrt(128)

  for (int kt = 0; kt < T_ / 32; ++kt) {
    const int key0 = kt * 32;

    // --- scores: two 16-key tiles, 4 d-chunks each -> 8 WMMAs ---
    v8f s0, s1;
#pragma unroll
    for (int i = 0; i < 8; ++i) { s0[i] = 0.0f; s1[i] = 0.0f; }
#pragma unroll
    for (int c = 0; c < 4; ++c) {
      // B 32x16 layout: lane holds key column (lane&15), elem e -> d = hl*16+e
      v16h kb0 = *(const v16h*)(Kb + (size_t)(key0 + ln16) * D_ + c * 32 + hl * 16);
      v16h kb1 = *(const v16h*)(Kb + (size_t)(key0 + 16 + ln16) * D_ + c * 32 + hl * 16);
      s0 = __builtin_amdgcn_wmma_f32_16x16x32_f16(false, qa[c], false, kb0,
                                                  (short)0, s0, false, false);
      s1 = __builtin_amdgcn_wmma_f32_16x16x32_f16(false, qa[c], false, kb1,
                                                  (short)0, s1, false, false);
    }
    if (kt + 1 < T_ / 32) {   // uniform condition: EXEC stays all-ones
      __builtin_prefetch(Kb + (size_t)(key0 + 32 + ln16) * D_, 0, 0);
      __builtin_prefetch(Vb + (size_t)ln16 * T_ + key0 + 32, 0, 0);
    }

    // --- online softmax over 32 new columns ---
    // C/D layout: lane = column, reg i = row (hl*8 + i); rows of the two
    // halves never mix under xor masks 1/2/4/8.
    float alpha[8];
#pragma unroll
    for (int i = 0; i < 8; ++i) {
      float a0 = s0[i] * scale, a1 = s1[i] * scale;
      float mx = fmaxf(a0, a1);
      mx = fmaxf(mx, __shfl_xor(mx, 1, 32));
      mx = fmaxf(mx, __shfl_xor(mx, 2, 32));
      mx = fmaxf(mx, __shfl_xor(mx, 4, 32));
      mx = fmaxf(mx, __shfl_xor(mx, 8, 32));
      float mnew = fmaxf(m[i], mx);
      alpha[i] = __expf(m[i] - mnew);
      m[i] = mnew;
      float p0 = __expf(a0 - mnew);
      float p1 = __expf(a1 - mnew);
      float rs = p0 + p1;
      rs += __shfl_xor(rs, 1, 32);
      rs += __shfl_xor(rs, 2, 32);
      rs += __shfl_xor(rs, 4, 32);
      rs += __shfl_xor(rs, 8, 32);
      l[i] = l[i] * alpha[i] + rs;
      Pl[hl * 8 + i][ln16]      = (_Float16)p0;
      Pl[hl * 8 + i][16 + ln16] = (_Float16)p1;
    }
#pragma unroll
    for (int dt = 0; dt < 8; ++dt)
#pragma unroll
      for (int i = 0; i < 8; ++i) acc[dt][i] *= alpha[i];
    __syncthreads();

    // Re-shape P from C-layout to A-layout via LDS (16x32 f16 tile).
    v16h pa;
    {
      const _Float16* pr = &Pl[ln16][0];
      pa = *(const v16h*)(pr + hl * 16);
    }
    __syncthreads();

    // --- PV: 8 output d-tiles of 16 -> 8 WMMAs; V^T gives contiguous B frags
    #pragma unroll
    for (int dt = 0; dt < 8; ++dt) {
      v16h vb = *(const v16h*)(Vb + (size_t)(dt * 16 + ln16) * T_ + key0 + hl * 16);
      acc[dt] = __builtin_amdgcn_wmma_f32_16x16x32_f16(false, pa, false, vb,
                                                       (short)0, acc[dt], false, false);
    }
  }

  // --- epilogue: normalize, scatter into ctx[t][b][hh*128 + d] (f32) ---
#pragma unroll
  for (int i = 0; i < 8; ++i) l[i] = 1.0f / l[i];
#pragma unroll
  for (int dt = 0; dt < 8; ++dt) {
#pragma unroll
    for (int i = 0; i < 8; ++i) {
      int trow = rowblk * 16 + hl * 8 + i;
      int col  = hh * D_ + dt * 16 + ln16;
      ctx[((size_t)trow * B_ + b) * HD_ + col] = acc[dt][i] * l[i];
    }
  }
}

// ---------------- output: ctx quant-dequant + reverse a2a -------------------

__global__ void output_kernel(const float* __restrict__ ctx, const float* __restrict__ ms,
                              float* __restrict__ out) {
  int u = blockIdx.x * 256 + threadIdx.x;    // one thread per (j, j+512) pair
  int j = u & 511; int r = u >> 9;
  int b = r & (B_ - 1); int tt = r >> 1;     // tt in [0,2048)
  int w = tt >> 9; int s = tt & 511;
  float mn = ms[0], step = ms[1];
  const float* crow = ctx + ((size_t)tt * B_ + b) * HD_;
  float hi, lo;
  qdq_pair(crow[j], crow[j + 512], mn, step, hi, lo);
  float* orow = out + ((size_t)s * B_ + b) * (W_ * HD_) + w * HD_;
  orow[j]       = hi;
  orow[j + 512] = lo;
}

// ---------------------------------------------------------------------------

extern "C" void kernel_launch(void* const* d_in, const int* in_sizes, int n_in,
                              void* d_out, int out_size, void* d_ws, size_t ws_size,
                              hipStream_t stream) {
  const float* q = (const float*)d_in[0];
  const float* k = (const float*)d_in[1];
  const float* v = (const float*)d_in[2];
  // d_in[3..5] = is_prompt, current_seq_len, world_size: geometry is fixed
  // (S=512,B=2,H=32,D=128,W=4) and the prompt path makes them no-ops here.
  float* out = (float*)d_out;

  char* ws = (char*)d_ws;
  _Float16* Qh = (_Float16*)(ws + QH_OFF);
  _Float16* Kh = (_Float16*)(ws + KH_OFF);
  _Float16* Vt = (_Float16*)(ws + VT_OFF);
  float*    ctx = (float*)(ws + CTX_OFF);
  float*    res = (float*)(ws + RES_OFF);    // 4 x {mn, step}
  float*    pmin = (float*)(ws + PART_OFF);
  float*    pmax = pmin + NPART_;

  const float* srcs[3] = {q, k, v};
  for (int i = 0; i < 3; ++i) {
    minmax_part_kernel<<<NPART_, 256, 0, stream>>>(srcs[i], N_, pmin, pmax);
    minmax_final_kernel<<<1, 256, 0, stream>>>(pmin, pmax, NPART_, res + 2 * i);
  }

  const int npairs = N_ / 2;                 // 2,097,152
  qk_transform_kernel<<<npairs / 256, 256, 0, stream>>>(q, res + 0, Qh);
  qk_transform_kernel<<<npairs / 256, 256, 0, stream>>>(k, res + 2, Kh);
  v_transform_kernel<<<npairs / 256, 256, 0, stream>>>(v, res + 4, Vt);

  flash_attn_kernel<<<B_ * HH_ * (T_ / 16), 32, 0, stream>>>(Qh, Kh, Vt, ctx);

  minmax_part_kernel<<<NPART_, 256, 0, stream>>>(ctx, N_, pmin, pmax);
  minmax_final_kernel<<<1, 256, 0, stream>>>(pmin, pmax, NPART_, res + 6);

  output_kernel<<<npairs / 256, 256, 0, stream>>>(ctx, res + 6, out);
}
